// PointGenerator_23691039605464
// MI455X (gfx1250) — matrix operand order repaired and checked by reading, not compiled, not run on hardware
//
#include <hip/hip_runtime.h>

// ---------------- constants ----------------
constexpr long long SB  = 16;      // batch
constexpr long long SN  = 2048;    // points
constexpr long long SK  = 10;      // knn k
constexpr long long SC1 = 128;     // FIN
constexpr long long SC2 = 256;     // 2*FIN
constexpr long long SP  = SB * SN * SK;   // 327680 positions (b,n,kk)
constexpr long long SP5 = SB * SN * 5;    // 163840 positions (b,n,t)
constexpr long long SBN = SB * SN;        // 32768

// ---------------- vector types ----------------
typedef __attribute__((ext_vector_type(16))) __bf16         bf16x16;
typedef __attribute__((ext_vector_type(8)))  float          f32x8;
typedef __attribute__((ext_vector_type(8)))  unsigned short u16x8;

union ABFrag { u16x8 h[2]; bf16x16 v; };

// ---------------- helpers ----------------
__device__ __forceinline__ unsigned short f2bf(float f) {
  unsigned int u = __float_as_uint(f);
  unsigned int r = u + 0x7FFFu + ((u >> 16) & 1u);   // round-nearest-even
  return (unsigned short)(r >> 16);
}
__device__ __forceinline__ float bf2f(unsigned short h) {
  return __uint_as_float(((unsigned int)h) << 16);
}
__device__ __forceinline__ float lrelu(float v) { return v >= 0.0f ? v : 0.01f * v; }

// A fragment (16x32 bf16, M = lane%16): lanes 0-15 hold K = kb+{0..7,16..23},
// lanes 16-31 hold K = kb+{8..15,24..31}.  row = contiguous bf16 row.
__device__ __forceinline__ bf16x16 load_a_frag(const unsigned short* __restrict__ row, int kb, int hi) {
  ABFrag u;
  u.h[0] = *(const u16x8*)(row + kb + hi * 8);
  u.h[1] = *(const u16x8*)(row + kb + 16 + hi * 8);
  return u.v;
}
// B fragment (32x16 bf16, N = lane%16): lanes 0-15 K=kb+{0..7,16..23}, lanes 16-31 K=kb+{8..15,24..31}.
// col = contiguous bf16 vector of K values for this output column.
__device__ __forceinline__ bf16x16 load_b_frag(const unsigned short* __restrict__ col, int kb, int hi) {
  ABFrag u;
  const unsigned short* p = col + kb + hi * 8;
  u.h[0] = *(const u16x8*)(p);
  u.h[1] = *(const u16x8*)(p + 16);
  return u.v;
}

// ---------------- conversions ----------------
__global__ void k_f2bf(const float* __restrict__ src, unsigned short* __restrict__ dst, long long n) {
  long long t = (long long)blockIdx.x * blockDim.x + threadIdx.x;
  if (t < n) dst[t] = f2bf(src[t]);
}

// xbT[b][n][c] = x[b][c][n]  (bf16, A-matrix friendly)
__global__ void k_xbt(const float* __restrict__ x, unsigned short* __restrict__ xbT) {
  long long t = (long long)blockIdx.x * blockDim.x + threadIdx.x;
  if (t >= SB * SN * SC1) return;
  long long b = t / (SN * SC1);
  long long r = t % (SN * SC1);
  long long n = r / SC1, c = r % SC1;
  xbT[t] = f2bf(x[(b * SC1 + c) * SN + n]);
}

__global__ void k_sqn(const unsigned short* __restrict__ xbT, float* __restrict__ sq) {
  long long t = (long long)blockIdx.x * blockDim.x + threadIdx.x;
  if (t >= SB * SN) return;
  const unsigned short* row = xbT + t * SC1;
  float s = 0.0f;
  for (int c = 0; c < (int)SC1; ++c) { float v = bf2f(row[c]); s += v * v; }
  sq[t] = s;
}

// Wt[o][d*256+c] = Wi[o][c][0][d]   (im2col weight order, bf16)
__global__ void k_wt(const float* __restrict__ Wi, unsigned short* __restrict__ Wt) {
  long long t = (long long)blockIdx.x * blockDim.x + threadIdx.x;
  if (t >= 512LL * 1536LL) return;
  long long o = t / 1536, r = t % 1536;
  long long d = r / 256, c = r % 256;
  Wt[t] = f2bf(Wi[o * 1536 + c * 6 + d]);
}

// ---------------- kNN: bf16-WMMA distance + top-10 ----------------
__global__ void __launch_bounds__(32) k_knn(const unsigned short* __restrict__ xbT,
                                            const float* __restrict__ sq,
                                            int* __restrict__ idx) {
  const int NT = (int)(SN / 16);            // 128 col tiles
  int b  = blockIdx.x / NT;
  int rt = blockIdx.x % NT;
  int lane = threadIdx.x, hi = lane >> 4, l16 = lane & 15;
  const unsigned short* xb = xbT + (long long)b * SN * SC1;
  const float* sqb = sq + (long long)b * SN;
  int rowg = rt * 16 + l16;

  bf16x16 afrag[4];
  #pragma unroll
  for (int s = 0; s < 4; ++s) afrag[s] = load_a_frag(xb + (long long)rowg * SC1, s * 32, hi);

  float tv[10]; int ti[10];
  #pragma unroll
  for (int q = 0; q < 10; ++q) { tv[q] = 3.0e38f; ti[q] = 0; }

  __shared__ float tile[16][17];

  for (int ct = 0; ct < NT; ++ct) {
    f32x8 acc = {};
    #pragma unroll
    for (int s = 0; s < 4; ++s) {
      bf16x16 bfr = load_b_frag(xb + (long long)(ct * 16 + l16) * SC1, s * 32, hi);
      acc = __builtin_amdgcn_wmma_f32_16x16x32_bf16(false, afrag[s], false, bfr,
                                                    (short)0, acc, false, false);
    }
    float sqc = sqb[ct * 16 + l16];
    #pragma unroll
    for (int v = 0; v < 8; ++v) {
      int rl = v + hi * 8;
      tile[rl][l16] = sqb[rt * 16 + rl] + sqc - 2.0f * acc[v];
    }
    __syncthreads();
    if (lane < 16) {
      for (int j = 0; j < 16; ++j) {
        int colg = ct * 16 + j;
        if (colg == rowg) continue;            // drop self (argsort [1:k+1])
        float dv = tile[lane][j]; int di = colg;
        #pragma unroll
        for (int q = 0; q < 10; ++q) {
          if (dv < tv[q]) { float t0 = tv[q]; int i0 = ti[q]; tv[q] = dv; ti[q] = di; dv = t0; di = i0; }
        }
      }
    }
    __syncthreads();
  }
  if (lane < 16) {
    #pragma unroll
    for (int q = 0; q < 10; ++q) idx[((long long)b * SN + rowg) * SK + q] = ti[q];
  }
}

// ---------------- e_fea / e_xyz build ----------------
__global__ void k_build(const unsigned short* __restrict__ xbT, const float* __restrict__ pc,
                        const int* __restrict__ idx, unsigned short* __restrict__ afea,
                        float* __restrict__ exyz) {
  long long p = blockIdx.x;           // (b,n,kk)
  int c = threadIdx.x;                // 0..255
  long long bn = p / SK;
  long long b = bn / SN, n = bn % SN;
  int j = idx[p];
  unsigned short v;
  if (c < (int)SC1) v = xbT[bn * SC1 + c];
  else {
    int cc = c - (int)SC1;
    v = f2bf(bf2f(xbT[((long long)b * SN + j) * SC1 + cc]) - bf2f(xbT[bn * SC1 + cc]));
  }
  afea[p * SC2 + c] = v;
  if (c < 6) {
    float e;
    if (c < 3) e = pc[(b * 3 + c) * SN + n];
    else { int d = c - 3; e = pc[(b * 3 + d) * SN + j] - pc[(b * 3 + d) * SN + n]; }
    exyz[p * 6 + c] = e;
  }
}

// ---------------- generic bf16 WMMA GEMM: C[M,Nn] = A[M,K] * W[Nn,K]^T ----------------
// N-tile blocked: each wave computes a 16 x (16*NT) output strip so the A fragment
// is reused NT times per K-step (A traffic / NT).
// A row byte layout indexed via: row_start(m) = (m/a_div)*a_mul + (m%a_div)*a_sub
// (normal GEMM: a_div=1, a_mul=Kd, a_sub=0; interp conv im2col: a_div=5, a_mul=2560, a_sub=256)
template <int NT>
__global__ void __launch_bounds__(32) k_gemm_bf16(const unsigned short* __restrict__ A,
                                                  const unsigned short* __restrict__ W,
                                                  float* __restrict__ C,
                                                  int Nn, int Kd,
                                                  int a_div, long long a_mul, long long a_sub) {
  int lane = threadIdx.x, hi = lane >> 4, l16 = lane & 15;
  long long m = (long long)blockIdx.x * 16 + l16;
  const unsigned short* arow = A + (m / a_div) * a_mul + (m % a_div) * a_sub;
  const unsigned short* wrow[NT];
  #pragma unroll
  for (int nt = 0; nt < NT; ++nt)
    wrow[nt] = W + ((long long)blockIdx.y * (16 * NT) + nt * 16 + l16) * (long long)Kd;

  f32x8 acc[NT] = {};
  for (int kb = 0; kb < Kd; kb += 32) {
    bf16x16 av = load_a_frag(arow, kb, hi);
    if (kb + 128 < Kd) __builtin_prefetch(arow + kb + 128, 0, 3);  // global_prefetch_b8
    #pragma unroll
    for (int nt = 0; nt < NT; ++nt) {
      bf16x16 bv = load_b_frag(wrow[nt], kb, hi);
      acc[nt] = __builtin_amdgcn_wmma_f32_16x16x32_bf16(false, av, false, bv,
                                                        (short)0, acc[nt], false, false);
    }
  }
  long long rbase = (long long)blockIdx.x * 16 + hi * 8;
  #pragma unroll
  for (int nt = 0; nt < NT; ++nt) {
    long long ncol = (long long)blockIdx.y * (16 * NT) + nt * 16 + l16;
    #pragma unroll
    for (int v = 0; v < 8; ++v) C[(rbase + v) * Nn + ncol] = acc[nt][v];
  }
}

// ---------------- tiny-K VALU stages ----------------
__global__ void k_xproj(const float* __restrict__ exyz, const float* __restrict__ Wx,
                        float* __restrict__ h16x) {
  long long p = (long long)blockIdx.x * blockDim.x + threadIdx.x;
  if (p >= SP) return;
  float e[6];
  #pragma unroll
  for (int i = 0; i < 6; ++i) e[i] = exyz[p * 6 + i];
  #pragma unroll
  for (int o = 0; o < 16; ++o) {
    const float* wr = Wx + o * 6;
    float s = 0.0f;
    #pragma unroll
    for (int i = 0; i < 6; ++i) s += e[i] * wr[i];
    h16x[p * 16 + o] = s;
  }
}

__global__ void k_a1(const float* __restrict__ w16, const float* __restrict__ Wa1,
                     float* __restrict__ h64) {
  long long t = (long long)blockIdx.x * blockDim.x + threadIdx.x;
  if (t >= SP * 64) return;
  long long p = t >> 6; int o = (int)(t & 63);
  const float* wr = Wa1 + o * 16;
  const float* xr = w16 + p * 16;
  float s = 0.0f;
  #pragma unroll
  for (int c = 0; c < 16; ++c) s += xr[c] * wr[c];
  h64[t] = s;
}

// ---------------- BN statistics (one block per channel, no atomics) ----------------
__global__ void k_stats(const float* __restrict__ src, int C, long long M,
                        float* __restrict__ sum, float* __restrict__ sq) {
  int c = blockIdx.x, t = threadIdx.x;
  float s = 0.0f, q = 0.0f;
  for (long long r = t; r < M; r += 256) {
    float v = src[r * C + c];
    s += v; q += v * v;
  }
  __shared__ float ls[256], lq[256];
  ls[t] = s; lq[t] = q;
  __syncthreads();
  for (int o = 128; o > 0; o >>= 1) {
    if (t < o) { ls[t] += ls[t + o]; lq[t] += lq[t + o]; }
    __syncthreads();
  }
  if (t == 0) { sum[c] = ls[0]; sq[c] = lq[0]; }
}

__global__ void k_finalize(const float* __restrict__ sum, const float* __restrict__ sq,
                           float* __restrict__ mean, float* __restrict__ rstd,
                           int C, float invM) {
  int c = blockIdx.x * blockDim.x + threadIdx.x;
  if (c >= C) return;
  float m = sum[c] * invM;
  float v = sq[c] * invM - m * m;
  mean[c] = m;
  rstd[c] = rsqrtf(v + 1e-5f);
}

// ---------------- BN / activation epilogues ----------------
// w16 = lrelu(bnF(h16f)) * lrelu(bnX(h16x)); written in place over h16f
__global__ void k_w16(float* __restrict__ h16f, const float* __restrict__ h16x,
                      const float* mf, const float* rf, const float* gf, const float* bf_,
                      const float* mx, const float* rx, const float* gx, const float* bx_) {
  long long t = (long long)blockIdx.x * blockDim.x + threadIdx.x;
  if (t >= SP * 16) return;
  int c = (int)(t & 15);
  float a = lrelu((h16f[t] - mf[c]) * rf[c] * gf[c] + bf_[c]);
  float b = lrelu((h16x[t] - mx[c]) * rx[c] * gx[c] + bx_[c]);
  h16f[t] = a * b;
}

__global__ void k_bnact(const float* __restrict__ h64, unsigned short* __restrict__ w64b,
                        const float* m, const float* r, const float* g, const float* bta) {
  long long t = (long long)blockIdx.x * blockDim.x + threadIdx.x;
  if (t >= SP * 64) return;
  int c = (int)(t & 63);
  w64b[t] = f2bf(lrelu((h64[t] - m[c]) * r[c] * g[c] + bta[c]));
}

// bn_a2 + lrelu + softmax over k axis, in place in h256 [bn][kk][c]
__global__ void k_softmax(float* __restrict__ h256,
                          const float* m, const float* r, const float* g, const float* bta) {
  long long t = (long long)blockIdx.x * blockDim.x + threadIdx.x;
  if (t >= SBN * 256) return;
  long long bn = t >> 8; int c = (int)(t & 255);
  float mm = m[c], rr = r[c], gg = g[c], bb = bta[c];
  float z[10]; float mx = -3.0e38f;
  #pragma unroll
  for (int kk = 0; kk < 10; ++kk) {
    float v = h256[(bn * SK + kk) * 256 + c];
    v = lrelu((v - mm) * rr * gg + bb);
    z[kk] = v;
    mx = fmaxf(mx, v);
  }
  float s = 0.0f;
  #pragma unroll
  for (int kk = 0; kk < 10; ++kk) { z[kk] = __expf(z[kk] - mx); s += z[kk]; }
  float inv = 1.0f / s;
  #pragma unroll
  for (int kk = 0; kk < 10; ++kk) h256[(bn * SK + kk) * 256 + c] = z[kk] * inv;
}

// merged A matrix for conv2: Am[bn][c*20+w]; w<10 -> e_fea, w>=10 -> lrelu(bn_i(inte))*softmax_w
// channel-pair interleave: (c,kk) <-> (o,t) with o=(c*10+kk)/5, t=(c*10+kk)%5
__global__ void k_merge(const unsigned short* __restrict__ afea, const float* __restrict__ inte,
                        const float* __restrict__ wsoft, unsigned short* __restrict__ Am,
                        const float* mi, const float* ri, const float* gi, const float* bi) {
  long long t = (long long)blockIdx.x * blockDim.x + threadIdx.x;
  if (t >= SBN * 5120) return;
  int w = (int)(t % 20);
  long long rest = t / 20;
  int c = (int)(rest % 256);
  long long bn = rest / 256;
  unsigned short outv;
  if (w < 10) {
    outv = afea[(bn * SK + w) * SC2 + c];
  } else {
    int kk = w - 10;
    int flat = c * 10 + kk;
    int o = flat / 5, tt = flat % 5;
    float iv = inte[(bn * 5 + tt) * 512 + o];
    iv = lrelu((iv - mi[o]) * ri[o] * gi[o] + bi[o]);
    float wv = wsoft[(bn * SK + kk) * 256 + c];
    outv = f2bf(iv * wv);
  }
  Am[t] = outv;
}

// final bn2 + relu + transpose [bn][o] -> out[b][o][n] (== [B, Fout, 2N] reshape)
__global__ void k_out(const float* __restrict__ outp, float* __restrict__ out,
                      const float* m, const float* r, const float* g, const float* bta) {
  long long t = (long long)blockIdx.x * blockDim.x + threadIdx.x;
  if (t >= SBN * 256) return;
  long long bn = t >> 8; int o = (int)(t & 255);
  long long b = bn / SN, n = bn % SN;
  float v = (outp[t] - m[o]) * r[o] * g[o] + bta[o];
  out[(b * 256 + o) * SN + n] = fmaxf(v, 0.0f);
}

// ---------------- host orchestration ----------------
extern "C" void kernel_launch(void* const* d_in, const int* in_sizes, int n_in,
                              void* d_out, int out_size, void* d_ws, size_t ws_size,
                              hipStream_t stream) {
  (void)in_sizes; (void)n_in; (void)out_size; (void)ws_size;
  const float* x     = (const float*)d_in[0];
  const float* pc    = (const float*)d_in[1];
  const float* Wf    = (const float*)d_in[2];
  const float* gf    = (const float*)d_in[4];
  const float* betaf = (const float*)d_in[5];
  const float* Wx    = (const float*)d_in[6];
  const float* gx    = (const float*)d_in[8];
  const float* betax = (const float*)d_in[9];
  const float* Wa1   = (const float*)d_in[10];
  const float* ga1   = (const float*)d_in[12];
  const float* bta1  = (const float*)d_in[13];
  const float* Wa2   = (const float*)d_in[14];
  const float* ga2   = (const float*)d_in[16];
  const float* bta2  = (const float*)d_in[17];
  const float* Wi    = (const float*)d_in[18];
  const float* gi    = (const float*)d_in[20];
  const float* btai  = (const float*)d_in[21];
  const float* W2    = (const float*)d_in[22];
  const float* g2    = (const float*)d_in[24];
  const float* bta22 = (const float*)d_in[25];
  // conv biases (d_in[3,7,11,15,19,23]) cancel exactly under training-mode BN -> skipped.
  float* out = (float*)d_out;

  // workspace carve-out (256B aligned)
  char* ws = (char*)d_ws;
  size_t off = 0;
  auto carve = [&](size_t bytes) -> char* {
    char* p = ws + off;
    off += (bytes + 255) & ~(size_t)255;
    return p;
  };
  unsigned short* xbT  = (unsigned short*)carve(SB * SN * SC1 * 2);    // 8.4 MB
  float*          sqn  = (float*)carve(SB * SN * 4);
  int*            idx  = (int*)carve(SP * 4);
  unsigned short* afea = (unsigned short*)carve(SP * SC2 * 2);         // 168 MB
  float*          exyz = (float*)carve(SP * 6 * 4);
  float*          h16f = (float*)carve(SP * 16 * 4);
  float*          h16x = (float*)carve(SP * 16 * 4);
  float*          h64  = (float*)carve(SP * 64 * 4);
  unsigned short* w64b = (unsigned short*)carve(SP * 64 * 2);
  float*          h256 = (float*)carve(SP * 256 * 4);                  // 336 MB
  unsigned short* wt   = (unsigned short*)carve(512 * 1536 * 2);
  float*          inte = (float*)carve(SP5 * 512 * 4);                 // 336 MB
  unsigned short* am   = (unsigned short*)carve(SBN * 5120 * 2);       // 336 MB
  float*          outp = (float*)carve(SBN * 256 * 4);
  unsigned short* wfb  = (unsigned short*)carve(16 * 256 * 2);
  unsigned short* wa2b = (unsigned short*)carve(256 * 64 * 2);
  unsigned short* w2b  = (unsigned short*)carve(256 * 5120 * 2);
  // BN stats: sum, sumsq, mean, rstd per layer
  auto stat4 = [&](int C, float** s, float** q, float** m, float** r) {
    *s = (float*)carve(C * 4); *q = (float*)carve(C * 4);
    *m = (float*)carve(C * 4); *r = (float*)carve(C * 4);
  };
  float *sF,*qF,*mF,*rF,  *sX,*qX,*mX,*rX,  *sA1,*qA1,*mA1,*rA1;
  float *sA2,*qA2,*mA2,*rA2,  *sI,*qI,*mI,*rI,  *s2,*q2,*m2,*r2;
  stat4(16,&sF,&qF,&mF,&rF);   stat4(16,&sX,&qX,&mX,&rX);
  stat4(64,&sA1,&qA1,&mA1,&rA1); stat4(256,&sA2,&qA2,&mA2,&rA2);
  stat4(512,&sI,&qI,&mI,&rI);  stat4(256,&s2,&q2,&m2,&r2);

  auto g1 = [](long long n) { return dim3((unsigned)((n + 255) / 256)); };
  const dim3 b256(256);

  // 0) conversions
  k_f2bf<<<g1(16 * 256), b256, 0, stream>>>(Wf, wfb, 16 * 256);
  k_f2bf<<<g1(256 * 64), b256, 0, stream>>>(Wa2, wa2b, 256 * 64);
  k_f2bf<<<g1(256 * 5120), b256, 0, stream>>>(W2, w2b, 256 * 5120);
  k_wt<<<g1(512 * 1536), b256, 0, stream>>>(Wi, wt);
  k_xbt<<<g1(SB * SN * SC1), b256, 0, stream>>>(x, xbT);
  k_sqn<<<g1(SB * SN), b256, 0, stream>>>(xbT, sqn);

  // 1) kNN via WMMA distance tiles
  k_knn<<<dim3((unsigned)(SB * (SN / 16))), dim3(32), 0, stream>>>(xbT, sqn, idx);

  // 2) e_fea / e_xyz
  k_build<<<dim3((unsigned)SP), b256, 0, stream>>>(xbT, pc, idx, afea, exyz);

  // 3) Wf branch: WMMA GEMM [P,256]x[16,256]^T ; Wx branch: K=6 VALU
  k_gemm_bf16<1><<<dim3((unsigned)(SP / 16), 1), dim3(32), 0, stream>>>(
      afea, wfb, h16f, 16, 256, 1, 256LL, 0LL);
  k_xproj<<<g1(SP), b256, 0, stream>>>(exyz, Wx, h16x);
  k_stats<<<dim3(16), b256, 0, stream>>>(h16f, 16, SP, sF, qF);
  k_stats<<<dim3(16), b256, 0, stream>>>(h16x, 16, SP, sX, qX);
  k_finalize<<<dim3(1), b256, 0, stream>>>(sF, qF, mF, rF, 16, 1.0f / (float)SP);
  k_finalize<<<dim3(1), b256, 0, stream>>>(sX, qX, mX, rX, 16, 1.0f / (float)SP);
  k_w16<<<g1(SP * 16), b256, 0, stream>>>(h16f, h16x, mF, rF, gf, betaf, mX, rX, gx, betax);

  // 4) Wa1 (K=16, VALU) -> BN -> lrelu -> bf16
  k_a1<<<g1(SP * 64), b256, 0, stream>>>(h16f, Wa1, h64);
  k_stats<<<dim3(64), b256, 0, stream>>>(h64, 64, SP, sA1, qA1);
  k_finalize<<<dim3(1), b256, 0, stream>>>(sA1, qA1, mA1, rA1, 64, 1.0f / (float)SP);
  k_bnact<<<g1(SP * 64), b256, 0, stream>>>(h64, w64b, mA1, rA1, ga1, bta1);

  // 5) Wa2: WMMA GEMM [P,64]x[256,64]^T (16x64 strips) -> BN+lrelu+softmax (in place)
  k_gemm_bf16<4><<<dim3((unsigned)(SP / 16), 4), dim3(32), 0, stream>>>(
      w64b, wa2b, h256, 256, 64, 1, 64LL, 0LL);
  k_stats<<<dim3(256), b256, 0, stream>>>(h256, 256, SP, sA2, qA2);
  k_finalize<<<dim3(1), b256, 0, stream>>>(sA2, qA2, mA2, rA2, 256, 1.0f / (float)SP);
  k_softmax<<<g1(SBN * 256), b256, 0, stream>>>(h256, mA2, rA2, ga2, bta2);

  // 6) interpolation conv = im2col WMMA GEMM [P5,1536]x[512,1536]^T (16x64 strips)
  //    A row for m=(bn,t): offset (bn*10+t)*256, contiguous 6*256 window -> a_div=5,a_mul=2560,a_sub=256
  k_gemm_bf16<4><<<dim3((unsigned)(SP5 / 16), 8), dim3(32), 0, stream>>>(
      afea, wt, inte, 512, 1536, 5, 2560LL, 256LL);
  k_stats<<<dim3(512), b256, 0, stream>>>(inte, 512, SP5, sI, qI);
  k_finalize<<<dim3(2), b256, 0, stream>>>(sI, qI, mI, rI, 512, 1.0f / (float)SP5);

  // 7) merge [e_fea | bn_i(inte)*w] -> Am[bn][5120]
  k_merge<<<g1(SBN * 5120), b256, 0, stream>>>(afea, inte, h256, am, mI, rI, gi, btai);

  // 8) conv2: WMMA GEMM [BN,5120]x[256,5120]^T (16x64 strips) -> BN -> relu -> transpose out
  k_gemm_bf16<4><<<dim3((unsigned)(SBN / 16), 4), dim3(32), 0, stream>>>(
      am, w2b, outp, 256, 5120, 1, 5120LL, 0LL);
  k_stats<<<dim3(256), b256, 0, stream>>>(outp, 256, SBN, s2, q2);
  k_finalize<<<dim3(1), b256, 0, stream>>>(s2, q2, m2, r2, 256, 1.0f / (float)SBN);
  k_out<<<g1(SBN * 256), b256, 0, stream>>>(outp, out, m2, r2, g2, bta22);
}
